// PerformerAttention_13718125543554
// MI455X (gfx1250) — matrix operand order repaired and checked by reading, not compiled, not run on hardware
//
#include <hip/hip_runtime.h>
#include <math.h>

typedef __bf16 bf16;
typedef __attribute__((ext_vector_type(16))) __bf16 v16bf;
typedef __attribute__((ext_vector_type(4)))  __bf16 bf16x4;
typedef __attribute__((ext_vector_type(8)))  float  v8f;
typedef __attribute__((ext_vector_type(4)))  int    v4i;

#define B_  4
#define L_  2048
#define DM_ 1024
#define H_  16
#define D_  64
#define M_  256
#define C_  128
#define NCH 16

#if defined(__has_builtin)
#  if __has_builtin(__builtin_amdgcn_global_load_async_to_lds_b128)
#    define HAVE_ASYNC_LDS 1
#  endif
#endif
#ifndef HAVE_ASYNC_LDS
#  define HAVE_ASYNC_LDS 0
#endif

#define AS1 __attribute__((address_space(1)))
#define AS3 __attribute__((address_space(3)))

// Async global->LDS 16B copy (ASYNCcnt path) with synchronous fallback.
// Builtin signature (from compiler diagnostic): (v4i32* global, v4i32* lds, imm offset, imm cpol)
static __device__ __forceinline__ void copy_b128_async(const uint4* g, uint4* s) {
#if HAVE_ASYNC_LDS
  __builtin_amdgcn_global_load_async_to_lds_b128((AS1 v4i*)g, (AS3 v4i*)s, 0, 0);
#else
  *s = *g;
#endif
}
static __device__ __forceinline__ void async_wait() {
#if HAVE_ASYNC_LDS
#  if __has_builtin(__builtin_amdgcn_s_wait_asynccnt)
  __builtin_amdgcn_s_wait_asynccnt(0);
#  else
  asm volatile("s_wait_asynccnt 0" ::: "memory");
#  endif
#endif
}

static __device__ __forceinline__ v8f wmma_bf16(v16bf a, v16bf b, v8f c) {
  return __builtin_amdgcn_wmma_f32_16x16x32_bf16(false, a, false, b, (short)0, c, false, false);
}

union FragU { uint4 u[2]; v16bf v; };
union PackU { uint4 u; bf16 h[8]; };

// A fragment (16x32, row-major source). Two contiguous 16B LDS loads per lane.
static __device__ __forceinline__ v16bf ldsA(const bf16* base, int ld, int row0, int k0) {
  int lane = threadIdx.x & 31;
  int r  = row0 + (lane & 15);
  int kh = k0 + ((lane >> 4) << 3);
  const bf16* p = base + r * ld + kh;
  FragU x;
  x.u[0] = *(const uint4*)(p);
  x.u[1] = *(const uint4*)(p + 16);
  return x.v;
}

// B fragment (32x16) from n-major (transposed) source: B[k][n] = base[(col0+n)*ld + k0+k]
static __device__ __forceinline__ v16bf ldsBT(const bf16* base, int ld, int k0, int col0) {
  int lane = threadIdx.x & 31;
  int n  = col0 + (lane & 15);
  int kh = k0 + ((lane >> 4) << 3);
  const bf16* p = base + n * ld + kh;
  FragU x;
  x.u[0] = *(const uint4*)(p);
  x.u[1] = *(const uint4*)(p + 16);
  return x.v;
}

// C/D f32 16x16 transposed store: out[(col)*ld + row] -> 8 contiguous bf16 = 1 b128 store
static __device__ __forceinline__ void store_tile_bf16_T(bf16* base, int ld, int row0, int col0, v8f c) {
  int lane = threadIdx.x & 31;
  int col = col0 + (lane & 15);
  int r0  = row0 + ((lane >> 4) << 3);
  PackU x;
#pragma unroll
  for (int i = 0; i < 8; i++) x.h[i] = (bf16)c[i];
  *(uint4*)(base + col * ld + r0) = x.u;
}

__global__ void k_init(unsigned* slot) { *slot = 0u; }

// ---------------- projection GEMM: out(bf16,(b,h,l,d)) = (X@W + bias)*mul ----------------
__global__ void __launch_bounds__(256) k_proj(const float* __restrict__ X,
                                              const float* __restrict__ W,
                                              const float* __restrict__ bias,
                                              bf16* __restrict__ out, float mul) {
  extern __shared__ char smem[];
  bf16* sA  = (bf16*)smem;        // 128x32 row-major
  bf16* sBt = sA + 128 * 32;      // 128(n) x 32(k) transposed
  const int tid  = threadIdx.x;
  const int wave = tid >> 5;
  const int lane = tid & 31;
  const int row0 = blockIdx.y * 128;
  const int col0 = blockIdx.x * 128;
  v8f acc[8];
#pragma unroll
  for (int c = 0; c < 8; c++) acc[c] = (v8f){0.f,0.f,0.f,0.f,0.f,0.f,0.f,0.f};

  for (int k0 = 0; k0 < DM_; k0 += 32) {
    for (int e = tid; e < 1024; e += 256) {          // X tile 128x32 as float4
      int idx = e * 4; int r = idx >> 5, c = idx & 31;
      float4 f = *(const float4*)(X + (size_t)(row0 + r) * DM_ + k0 + c);
      bf16x4 o = { (bf16)f.x, (bf16)f.y, (bf16)f.z, (bf16)f.w };
      *(bf16x4*)(sA + idx) = o;
    }
    for (int e = tid; e < 1024; e += 256) {          // W tile 32x128, coalesced reads
      int idx = e * 4; int r = idx >> 7, c = idx & 127;
      float4 f = *(const float4*)(W + (size_t)(k0 + r) * DM_ + col0 + c);
      sBt[(c + 0) * 32 + r] = (bf16)f.x;
      sBt[(c + 1) * 32 + r] = (bf16)f.y;
      sBt[(c + 2) * 32 + r] = (bf16)f.z;
      sBt[(c + 3) * 32 + r] = (bf16)f.w;
    }
    __syncthreads();
    v16bf a = ldsA(sA, 32, wave * 16, 0);
#pragma unroll
    for (int c = 0; c < 8; c++)
      acc[c] = wmma_bf16(a, ldsBT(sBt, 32, 0, c * 16), acc[c]);
    __syncthreads();
  }
  int nn = lane & 15;
  int rb = wave * 16 + ((lane >> 4) << 3);
#pragma unroll
  for (int c = 0; c < 8; c++) {
    int col = col0 + c * 16 + nn;
    float bv = bias[col];
    int hh = col >> 6, dd = col & 63;
#pragma unroll
    for (int i = 0; i < 8; i++) {
      int row = row0 + rb + i;
      int bb = row >> 11, l = row & 2047;
      out[(((size_t)bb * H_ + hh) * L_ + l) * D_ + dd] = (bf16)((acc[c][i] + bv) * mul);
    }
  }
}

// ---------------- output GEMM: out(f32) = ctx(bf16)@Wo + bo ----------------
__global__ void __launch_bounds__(256) k_outproj(const bf16* __restrict__ ctx,
                                                 const float* __restrict__ W,
                                                 const float* __restrict__ bias,
                                                 float* __restrict__ out) {
  extern __shared__ char smem[];
  bf16* sA  = (bf16*)smem;
  bf16* sBt = sA + 128 * 32;
  const int tid  = threadIdx.x;
  const int wave = tid >> 5;
  const int lane = tid & 31;
  const int row0 = blockIdx.y * 128;
  const int col0 = blockIdx.x * 128;
  v8f acc[8];
#pragma unroll
  for (int c = 0; c < 8; c++) acc[c] = (v8f){0.f,0.f,0.f,0.f,0.f,0.f,0.f,0.f};

  for (int k0 = 0; k0 < DM_; k0 += 32) {
    for (int e = tid; e < 512; e += 256) {           // 128x32 bf16, async 16B copies
      int idx = e * 8; int r = idx >> 5, c = idx & 31;
      copy_b128_async((const uint4*)(ctx + (size_t)(row0 + r) * DM_ + k0 + c),
                      (uint4*)(sA + idx));
    }
    for (int e = tid; e < 1024; e += 256) {
      int idx = e * 4; int r = idx >> 7, c = idx & 127;
      float4 f = *(const float4*)(W + (size_t)(k0 + r) * DM_ + col0 + c);
      sBt[(c + 0) * 32 + r] = (bf16)f.x;
      sBt[(c + 1) * 32 + r] = (bf16)f.y;
      sBt[(c + 2) * 32 + r] = (bf16)f.z;
      sBt[(c + 3) * 32 + r] = (bf16)f.w;
    }
    async_wait();
    __syncthreads();
    v16bf a = ldsA(sA, 32, wave * 16, 0);
#pragma unroll
    for (int c = 0; c < 8; c++)
      acc[c] = wmma_bf16(a, ldsBT(sBt, 32, 0, c * 16), acc[c]);
    __syncthreads();
  }
  int nn = lane & 15;
  int rb = wave * 16 + ((lane >> 4) << 3);
#pragma unroll
  for (int c = 0; c < 8; c++) {
    int col = col0 + c * 16 + nn;
    float bv = bias[col];
#pragma unroll
    for (int i = 0; i < 8; i++)
      out[(size_t)(row0 + rb + i) * DM_ + col] = acc[c][i] + bv;
  }
}

// hk = -0.5*||ks_row||^2 ; global max via order-preserving uint encoding
__global__ void __launch_bounds__(256) k_hk(const bf16* __restrict__ ks,
                                            float* __restrict__ hk, unsigned* slot) {
  int t = blockIdx.x * 256 + threadIdx.x;
  const v16bf* p = (const v16bf*)(ks + (size_t)t * D_);
  float s = 0.f;
#pragma unroll
  for (int j = 0; j < 4; j++) {
    v16bf x = p[j];
#pragma unroll
    for (int i = 0; i < 16; i++) { float f = (float)x[i]; s += f * f; }
  }
  float h = -0.5f * s;
  hk[t] = h;
  unsigned u   = __float_as_uint(h);
  unsigned key = (u & 0x80000000u) ? ~u : (u | 0x80000000u);
  __shared__ unsigned red[256];
  red[threadIdx.x] = key;
  __syncthreads();
  for (int off = 128; off > 0; off >>= 1) {
    if (threadIdx.x < off) red[threadIdx.x] = max(red[threadIdx.x], red[threadIdx.x + off]);
    __syncthreads();
  }
  if (threadIdx.x == 0) atomicMax(slot, red[0]);
}

// ---------------- chunked causal linear attention, one block per (b,h) ----------------
__global__ void __launch_bounds__(256, 1) k_attn(const bf16* __restrict__ qs,
                                                 const bf16* __restrict__ ks,
                                                 const bf16* __restrict__ vh,
                                                 const float* __restrict__ RF,
                                                 const float* __restrict__ hkg,
                                                 const unsigned* __restrict__ slot,
                                                 bf16* __restrict__ ctx) {
  extern __shared__ char smem[];
  bf16* sRFt = (bf16*)smem;            // 256(m) x 64(d)   transposed RF
  bf16* sQ   = sRFt + 256 * 64;        // 128 x 256        qc row-major
  bf16* sK   = sQ + 128 * 256;         // 128 x 256        kc row-major (j-major)
  bf16* sKt  = sK + 128 * 256;         // 256(m) x 128(j)  kc m-major
  bf16* sVt  = sKt + 256 * 128;        // 64(d) x 128(j)   vc d-major
  bf16* sAX  = sVt + 64 * 128;         // 128x128 scores sA  (aliases 128x64 sX)
  bf16* sSt  = sAX + 128 * 128;        // 64(d) x 256(m)   Sprev transposed
  float* sZ    = (float*)(sSt + 64 * 256); // 256  zprev
  float* sZacc = sZ + 256;                 // 256  running z (ds_add_f32)
  float* sDen  = sZacc + 256;              // 128
  float* sHk   = sDen + 128;               // 128
  bf16* sA = sAX;
  bf16* sX = sAX;                       // phase-disjoint alias

  const int tid  = threadIdx.x;
  const int wave = tid >> 5;
  const int lane = tid & 31;
  const int bh = blockIdx.x;
  const int b  = bh >> 4, h = bh & 15;

  unsigned key = *slot;
  unsigned u = (key & 0x80000000u) ? (key ^ 0x80000000u) : ~key;
  const float kstab = __uint_as_float(u);

  // stage RF transposed (coalesced read, strided LDS write)
  for (int e = tid; e < 64 * 256; e += 256) {
    int d = e >> 8, m = e & 255;
    sRFt[m * 64 + d] = (bf16)RF[e];
  }
  sZacc[tid] = 0.f;

  v8f Sacc[2][4];                       // wave owns m rows [wave*32, wave*32+32)
#pragma unroll
  for (int rt = 0; rt < 2; rt++)
#pragma unroll
    for (int c = 0; c < 4; c++) Sacc[rt][c] = (v8f){0.f,0.f,0.f,0.f,0.f,0.f,0.f,0.f};
  __syncthreads();

  const size_t rowbase = (size_t)bh * L_;
  const int nn = lane & 15;
  const int rb = wave * 16 + ((lane >> 4) << 3);

  for (int n = 0; n < NCH; n++) {
    const int l0 = n * C_;
    if (n + 1 < NCH) {                  // prefetch next chunk (global_prefetch_b8)
      const bf16* pq = qs + (rowbase + l0 + C_) * D_ + (size_t)tid * 32;
      const bf16* pk = ks + (rowbase + l0 + C_) * D_ + (size_t)tid * 32;
      const bf16* pv = vh + (rowbase + l0 + C_) * D_ + (size_t)tid * 32;
      __builtin_prefetch(pq, 0, 0);
      __builtin_prefetch(pk, 0, 0);
      __builtin_prefetch(pv, 0, 0);
    }
    // ---- a: stage q-chunk (sX, async) + v-chunk (sVt, transposed); snapshot Sprev/zprev ----
    {
      const uint4* g = (const uint4*)(qs + (rowbase + l0) * D_);
      uint4* sx = (uint4*)sX;
      for (int e = tid; e < 1024; e += 256) copy_b128_async(g + e, sx + e);
      const bf16* gv = vh + (rowbase + l0) * D_;
      for (int e8 = tid; e8 < 1024; e8 += 256) {     // j = e8>>3, d0 = (e8&7)*8
        int j = e8 >> 3, d0 = (e8 & 7) * 8;
        PackU x; x.u = *(const uint4*)(gv + (size_t)j * D_ + d0);
#pragma unroll
        for (int i = 0; i < 8; i++) sVt[(d0 + i) * C_ + j] = x.h[i];
      }
      sZ[tid] = sZacc[tid];
#pragma unroll
      for (int rt = 0; rt < 2; rt++)
#pragma unroll
        for (int c = 0; c < 4; c++)
          store_tile_bf16_T(sSt, M_, wave * 32 + rt * 16, c * 16, Sacc[rt][c]);
      if (tid < C_) sHk[tid] = hkg[rowbase + l0 + tid];
    }
    async_wait();
    __syncthreads();
    // ---- b: qc = c*(exp(X@RF)+eps) -> sQ ----
    {
      v16bf a0 = ldsA(sX, D_, wave * 16, 0);
      v16bf a1 = ldsA(sX, D_, wave * 16, 32);
      for (int ct = 0; ct < 16; ct++) {
        v8f acc = (v8f){0.f,0.f,0.f,0.f,0.f,0.f,0.f,0.f};
        acc = wmma_bf16(a0, ldsBT(sRFt, D_, 0, ct * 16), acc);
        acc = wmma_bf16(a1, ldsBT(sRFt, D_, 32, ct * 16), acc);
        int col = ct * 16 + nn;
#pragma unroll
        for (int i = 0; i < 8; i++)
          sQ[(rb + i) * M_ + col] = (bf16)(0.0625f * (__expf(acc[i]) + 1e-4f));
      }
    }
    __syncthreads();
    // ---- c: stage k-chunk into sX (async) ----
    {
      const uint4* g = (const uint4*)(ks + (rowbase + l0) * D_);
      uint4* sx = (uint4*)sX;
      for (int e = tid; e < 1024; e += 256) copy_b128_async(g + e, sx + e);
    }
    async_wait();
    __syncthreads();
    // ---- d: kc = c*(exp(hk + X@RF - kstab)+eps) -> sK, sKt ; z += partials (ds_add_f32) ----
    {
      v16bf a0 = ldsA(sX, D_, wave * 16, 0);
      v16bf a1 = ldsA(sX, D_, wave * 16, 32);
      for (int ct = 0; ct < 16; ct++) {
        v8f acc = (v8f){0.f,0.f,0.f,0.f,0.f,0.f,0.f,0.f};
        acc = wmma_bf16(a0, ldsBT(sRFt, D_, 0, ct * 16), acc);
        acc = wmma_bf16(a1, ldsBT(sRFt, D_, 32, ct * 16), acc);
        int col = ct * 16 + nn;                       // m index
        PackU x;
        float psum = 0.f;
#pragma unroll
        for (int i = 0; i < 8; i++) {
          float v = 0.0625f * (__expf(sHk[rb + i] + acc[i] - kstab) + 1e-4f);
          sK[(rb + i) * M_ + col] = (bf16)v;
          x.h[i] = (bf16)v;
          psum += v;
        }
        *(uint4*)(sKt + col * C_ + rb) = x.u;         // m-major packed copy
        atomicAdd(&sZacc[col], psum);
      }
    }
    __syncthreads();
    // ---- e: A = tril(qc @ kc^T) -> sA ----
    {
      for (int ct = 0; ct < 8; ct++) {
        int col = ct * 16 + nn;
        if (ct > wave) {                              // fully masked (wave-uniform)
#pragma unroll
          for (int i = 0; i < 8; i++) sA[(rb + i) * C_ + col] = (bf16)0.f;
          continue;
        }
        v8f acc = (v8f){0.f,0.f,0.f,0.f,0.f,0.f,0.f,0.f};
        for (int mk = 0; mk < M_; mk += 32)
          acc = wmma_bf16(ldsA(sQ, M_, wave * 16, mk), ldsBT(sK, M_, mk, ct * 16), acc);
#pragma unroll
        for (int i = 0; i < 8; i++) {
          float v = acc[i];
          if (ct == wave && (rb + i) < col) v = 0.f;
          sA[(rb + i) * C_ + col] = (bf16)v;
        }
      }
    }
    __syncthreads();
    // ---- f: den = rowsum(A) + qc . zprev (vectorized LDS reads) ----
    if (tid < C_) {
      const v16bf* rowA = (const v16bf*)(sA + tid * C_);
      float rs = 0.f;
#pragma unroll
      for (int j = 0; j < 8; j++) {
        v16bf t = rowA[j];
#pragma unroll
        for (int i = 0; i < 16; i++) rs += (float)t[i];
      }
      const v16bf* rowQ = (const v16bf*)(sQ + tid * M_);
      float qz = 0.f;
#pragma unroll
      for (int j = 0; j < 16; j++) {
        v16bf t = rowQ[j];
#pragma unroll
        for (int i = 0; i < 16; i++) qz += (float)t[i] * sZ[j * 16 + i];
      }
      float den = rs + qz;
      if (fabsf(den) <= 1e-6f) den += 2e-6f;
      sDen[tid] = den;
    }
    __syncthreads();
    // ---- g: num = A@V + qc@Sprev ; write ctx ; S += kc^T @ V ----
    {
      v8f num[4];
#pragma unroll
      for (int c = 0; c < 4; c++) num[c] = (v8f){0.f,0.f,0.f,0.f,0.f,0.f,0.f,0.f};
      for (int j = 0; j < C_; j += 32) {
        v16bf a = ldsA(sA, C_, wave * 16, j);
#pragma unroll
        for (int c = 0; c < 4; c++)
          num[c] = wmma_bf16(a, ldsBT(sVt, C_, j, c * 16), num[c]);
      }
      for (int mk = 0; mk < M_; mk += 32) {
        v16bf a = ldsA(sQ, M_, wave * 16, mk);
#pragma unroll
        for (int c = 0; c < 4; c++)
          num[c] = wmma_bf16(a, ldsBT(sSt, M_, mk, c * 16), num[c]);
      }
#pragma unroll
      for (int c = 0; c < 4; c++) {
        int dd = c * 16 + nn;
#pragma unroll
        for (int i = 0; i < 8; i++) {
          int li = rb + i;
          float o = num[c][i] / sDen[li];
          ctx[((size_t)b * L_ + l0 + li) * DM_ + h * D_ + dd] = (bf16)o;
        }
      }
      // S += kc^T @ V : contiguous A-frags from m-major sKt
      for (int j = 0; j < C_; j += 32) {
        v16bf a0 = ldsA(sKt, C_, wave * 32, j);
        v16bf a1 = ldsA(sKt, C_, wave * 32 + 16, j);
#pragma unroll
        for (int c = 0; c < 4; c++) {
          v16bf bb = ldsBT(sVt, C_, j, c * 16);
          Sacc[0][c] = wmma_bf16(a0, bb, Sacc[0][c]);
          Sacc[1][c] = wmma_bf16(a1, bb, Sacc[1][c]);
        }
      }
    }
    __syncthreads();
  }
}

extern "C" void kernel_launch(void* const* d_in, const int* in_sizes, int n_in,
                              void* d_out, int out_size, void* d_ws, size_t ws_size,
                              hipStream_t stream) {
  const float* q  = (const float*)d_in[0];
  const float* k  = (const float*)d_in[1];
  const float* v  = (const float*)d_in[2];
  const float* Wq = (const float*)d_in[3];
  const float* bq = (const float*)d_in[4];
  const float* Wk = (const float*)d_in[5];
  const float* bk = (const float*)d_in[6];
  const float* Wv = (const float*)d_in[7];
  const float* bv = (const float*)d_in[8];
  const float* Wo = (const float*)d_in[9];
  const float* bo = (const float*)d_in[10];
  const float* RF = (const float*)d_in[11];
  float* out = (float*)d_out;

  char* ws = (char*)d_ws;
  bf16* qs  = (bf16*)(ws + 0);                 // 16 MB (b,h,l,d)
  bf16* ks  = (bf16*)(ws + 16777216);          // 16 MB
  bf16* vh  = (bf16*)(ws + 33554432);          // 16 MB
  bf16* ctx = (bf16*)(ws + 50331648);          // 16 MB (b,l,dm)
  float* hk = (float*)(ws + 67108864);         // 512 KB
  unsigned* slot = (unsigned*)(ws + 67633152); // 4 B

  const float invscale = 0.17677669529663687f; // 1024^(-1/4)
  const size_t smemGemm = (128 * 32 + 128 * 32) * sizeof(bf16); // 16 KB
  const size_t smemAttn = (256*64 + 128*256 + 128*256 + 256*128 + 64*128 + 128*128 + 64*256) * sizeof(bf16)
                        + (256 + 256 + 128 + 128) * sizeof(float); // ~307 KB

  k_init<<<dim3(1), dim3(1), 0, stream>>>(slot);
  k_proj<<<dim3(8, 64), dim3(256), smemGemm, stream>>>(q, Wq, bq, qs, invscale);
  k_proj<<<dim3(8, 64), dim3(256), smemGemm, stream>>>(k, Wk, bk, ks, invscale);
  k_proj<<<dim3(8, 64), dim3(256), smemGemm, stream>>>(v, Wv, bv, vh, 1.0f);
  k_hk<<<dim3((B_ * H_ * L_) / 256), dim3(256), 0, stream>>>(ks, hk, slot);
  k_attn<<<dim3(B_ * H_), dim3(256), smemAttn, stream>>>(qs, ks, vh, RF, hk, slot, ctx);
  k_outproj<<<dim3(8, 64), dim3(256), smemGemm, stream>>>(ctx, Wo, bo, out);
}
